// Query_Guided_Attention_Layer_1151051235527
// MI455X (gfx1250) — matrix operand order repaired and checked by reading, not compiled, not run on hardware
//
#include <hip/hip_runtime.h>
#include <cstdint>

typedef __attribute__((ext_vector_type(16))) __bf16 v16bf;
typedef __attribute__((ext_vector_type(8)))  float  v8f;

#define C_DIM   2048
#define S_DIM   32
#define N_IMG   40
#define Q_NUM   8
#define P_PAIR  256
#define NCLS    751
#define NCLS_PAD 768
#define CHW     (C_DIM * S_DIM)   // 65536

// LDS B-panel: 32 rows x 64B data, padded to 72B row stride (bank-conflict-free,
// 8B-aligned for ds_load_b64). Two buffers for double buffering.
#define BROW_BYTES 72
#define BPANEL_BYTES (32 * BROW_BYTES)   // 2304

union V16 {
    v16bf v;
    uint4 q[2];
    unsigned long long d[4];
    unsigned short h[16];
};

__device__ __forceinline__ v8f wmma_bf16(const V16& a, const V16& b, v8f c) {
    return __builtin_amdgcn_wmma_f32_16x16x32_bf16(false, a.v, false, b.v,
                                                   (short)0, c, false, false);
}

// CDNA5 async global->LDS copy (ASYNCcnt-tracked). LDS address = 32-bit offset
// in VDST VGPR; global address in VADDR pair (GV mode, SADDR=off).
__device__ __forceinline__ void asyncLoadB64(uint32_t ldsOff, const void* g) {
    asm volatile("global_load_async_to_lds_b64 %0, %1, off"
                 :: "v"(ldsOff), "v"(g) : "memory");
}
__device__ __forceinline__ void waitAsync0() {
    asm volatile("s_wait_asynccnt 0x0" ::: "memory");
}

// ---------------------------------------------------------------- converts
__global__ void k_f32_to_bf16(const float* __restrict__ src,
                              __bf16* __restrict__ dst, int n) {
    int i = blockIdx.x * blockDim.x + threadIdx.x;
    int stride = gridDim.x * blockDim.x;
    for (; i < n; i += stride) dst[i] = (__bf16)src[i];
}

__global__ void k_fc_transpose(const float* __restrict__ fcw,
                               __bf16* __restrict__ dst) {
    int i = blockIdx.x * blockDim.x + threadIdx.x;
    int stride = gridDim.x * blockDim.x;
    const int total = C_DIM * NCLS_PAD;
    for (; i < total; i += stride) {
        int c = i / NCLS_PAD, j = i % NCLS_PAD;
        float v = (j < NCLS) ? fcw[j * C_DIM + c] : 0.0f;
        dst[i] = (__bf16)v;
    }
}

// ---------------------------------------------------------------- GEMM: OUT = W @ X (per batch)
// B panel (32x32 bf16 per k-step) staged in LDS via async copy, double-buffered.
// MODE 0: theta/phi (+bias, bf16 out; galleries also store theta^T for attention)
// MODE 1: conv1 (relu, bf16 out)
// MODE 2: conv2 (f32 residual accumulate in place: OutF32 += acc)
template <int MODE>
__global__ void k_gemm(const __bf16* __restrict__ Wa, const __bf16* __restrict__ Wb,
                       const float* __restrict__ biasA, const float* __restrict__ biasB,
                       const __bf16* __restrict__ Xin,
                       __bf16* __restrict__ OutBf, float* __restrict__ OutF32,
                       __bf16* __restrict__ OutBfT) {
    __shared__ __align__(16) unsigned char smemB[2 * BPANEL_BYTES];
    int tid   = threadIdx.x;
    int wavei = tid >> 5, lane = tid & 31;
    int b  = blockIdx.x >> 2;        // batch (image or pair)
    int oq = blockIdx.x & 3;
    int og = oq * 8 + wavei;         // 0..31
    int o0 = og * 64;                // 4 o-tiles per wave

    const __bf16* W;
    const float*  bias = nullptr;
    if (MODE == 0) {
        W    = (b < Q_NUM) ? Wb : Wa;
        bias = (b < Q_NUM) ? biasB : biasA;
    } else {
        W = Wa;
    }
    const __bf16* X = Xin + (size_t)b * CHW;
    int nlo = lane & 15, hf = lane >> 4;

    // per-thread slice of the B-panel fill: 256 threads x 8B = 2KB
    int frow = tid >> 3, fseg = tid & 7;
    uint32_t ldsBase = (uint32_t)(uintptr_t)smemB;
    uint32_t fillLds = ldsBase + (uint32_t)(frow * BROW_BYTES + fseg * 8);

    // prologue: fill buffer 0 with k0 = 0
    asyncLoadB64(fillLds, X + (size_t)frow * S_DIM + fseg * 4);

    v8f acc[4][2];
#pragma unroll
    for (int i = 0; i < 4; ++i) { acc[i][0] = {}; acc[i][1] = {}; }

    for (int it = 0; it < 64; ++it) {
        int k0 = it * 32;
        waitAsync0();        // my slice of buf[it&1] landed in LDS
        __syncthreads();     // everyone's slice landed; prev reads of buf[(it+1)&1] done
        int cur = it & 1;
        if (it + 1 < 64) {
            asyncLoadB64(ldsBase + (uint32_t)(((it + 1) & 1) * BPANEL_BYTES) +
                             (uint32_t)(frow * BROW_BYTES + fseg * 8),
                         X + (size_t)(k0 + 32 + frow) * S_DIM + fseg * 4);
        }

        V16 b0, b1;
        const unsigned short* rb = (const unsigned short*)(smemB + cur * BPANEL_BYTES);
#pragma unroll
        for (int s = 0; s < 16; ++s) {
            int kk = (s < 8) ? (8 * hf + s) : (16 + 8 * hf + (s - 8));
            b0.h[s] = rb[kk * (BROW_BYTES / 2) + nlo];
            b1.h[s] = rb[kk * (BROW_BYTES / 2) + 16 + nlo];
        }

#pragma unroll
        for (int i = 0; i < 4; ++i) {
            V16 a;
            const __bf16* wr = W + (size_t)(o0 + 16 * i + nlo) * C_DIM + k0 + 8 * hf;
            a.q[0] = *(const uint4*)wr;
            a.q[1] = *(const uint4*)(wr + 16);
            acc[i][0] = wmma_bf16(a, b0, acc[i][0]);
            acc[i][1] = wmma_bf16(a, b1, acc[i][1]);
        }
    }

#pragma unroll
    for (int i = 0; i < 4; ++i) {
#pragma unroll
        for (int st = 0; st < 2; ++st) {
#pragma unroll
            for (int r = 0; r < 8; ++r) {
                int o = o0 + 16 * i + r + 8 * hf;
                int s = st * 16 + nlo;
                float v = acc[i][st][r];
                size_t idx = (size_t)b * CHW + (size_t)o * S_DIM + s;
                if (MODE == 0) {
                    v += bias[o];
                    OutBf[idx] = (__bf16)v;
                    if (b >= Q_NUM)   // theta^T (row-major [s][c]) for attention A-operand
                        OutBfT[(size_t)(b - Q_NUM) * CHW + (size_t)s * C_DIM + o] = (__bf16)v;
                } else if (MODE == 1) {
                    v = v > 0.0f ? v : 0.0f;
                    OutBf[idx] = (__bf16)v;
                } else {
                    OutF32[idx] += v;
                }
            }
        }
    }
}

// ---------------------------------------------------------------- attention: f = theta^T @ phi, att = sigmoid(max_t f / 32)
__global__ void k_attention(const __bf16* __restrict__ TP, const __bf16* __restrict__ TGt,
                            float* __restrict__ att) {
    int p    = (blockIdx.x * blockDim.x + threadIdx.x) >> 5;   // pair 0..255
    int lane = threadIdx.x & 31;
    int qi = p >> 5;
    const __bf16* TGr = TGt + (size_t)(p & 31) * CHW;   // theta^T of gallery (gi-8)
    const __bf16* PQ  = TP + (size_t)qi * CHW;          // phi of query
    int nlo = lane & 15, hf = lane >> 4;

    v8f acc[2][2];
    acc[0][0] = {}; acc[0][1] = {}; acc[1][0] = {}; acc[1][1] = {};

    for (int k0 = 0; k0 < C_DIM; k0 += 32) {
        V16 b0, b1;
        const unsigned short* pq = (const unsigned short*)PQ;
#pragma unroll
        for (int s = 0; s < 16; ++s) {
            int kk = (s < 8) ? (8 * hf + s) : (16 + 8 * hf + (s - 8));
            b0.h[s] = pq[(size_t)(k0 + kk) * S_DIM + nlo];
            b1.h[s] = pq[(size_t)(k0 + kk) * S_DIM + 16 + nlo];
        }
#pragma unroll
        for (int st = 0; st < 2; ++st) {
            V16 a;
            const __bf16* ar = TGr + (size_t)(st * 16 + nlo) * C_DIM + k0 + 8 * hf;
            a.q[0] = *(const uint4*)ar;
            a.q[1] = *(const uint4*)(ar + 16);
            acc[st][0] = wmma_bf16(a, b0, acc[st][0]);
            acc[st][1] = wmma_bf16(a, b1, acc[st][1]);
        }
    }

#pragma unroll
    for (int st = 0; st < 2; ++st) {
#pragma unroll
        for (int r = 0; r < 8; ++r) {
            float v = fmaxf(acc[st][0][r], acc[st][1][r]);
            v = fmaxf(v, __shfl_xor(v, 1, 32));
            v = fmaxf(v, __shfl_xor(v, 2, 32));
            v = fmaxf(v, __shfl_xor(v, 4, 32));
            v = fmaxf(v, __shfl_xor(v, 8, 32));
            if (nlo == 0) {
                int s = st * 16 + r + 8 * hf;
                att[p * S_DIM + s] = 1.0f / (1.0f + __expf(-v * (1.0f / 32.0f)));
            }
        }
    }
}

// ---------------------------------------------------------------- pre-MLP: y = x_g*(1+att); LN1 -> h (f32 + bf16)
__global__ void k_premlp(const float* __restrict__ x, const float* __restrict__ att,
                         const float* __restrict__ n1w, const float* __restrict__ n1b,
                         float* __restrict__ hF, __bf16* __restrict__ hB) {
    int p = blockIdx.x, tid = threadIdx.x;
    int gi = Q_NUM + (p & 31);
    const float* xg = x + (size_t)gi * CHW;
    __shared__ float sAtt[32];
    __shared__ float red[256], red2[256];
    if (tid < 32) sAtt[tid] = att[p * S_DIM + tid];
    __syncthreads();

    float sum = 0.0f, sq = 0.0f;
    for (int i = tid; i < CHW / 4; i += 256) {
        float4 xv = ((const float4*)xg)[i];
        int s4 = (i & 7) * 4;
        float y0 = xv.x * (1.0f + sAtt[s4]);
        float y1 = xv.y * (1.0f + sAtt[s4 + 1]);
        float y2 = xv.z * (1.0f + sAtt[s4 + 2]);
        float y3 = xv.w * (1.0f + sAtt[s4 + 3]);
        sum += y0 + y1 + y2 + y3;
        sq  += y0 * y0 + y1 * y1 + y2 * y2 + y3 * y3;
    }
    red[tid] = sum; red2[tid] = sq;
    __syncthreads();
    for (int off = 128; off > 0; off >>= 1) {
        if (tid < off) { red[tid] += red[tid + off]; red2[tid] += red2[tid + off]; }
        __syncthreads();
    }
    float mu  = red[0] * (1.0f / CHW);
    float var = red2[0] * (1.0f / CHW) - mu * mu;
    float rs  = rsqrtf(var + 1e-5f);

    for (int i = tid; i < CHW / 4; i += 256) {
        float4 xv = ((const float4*)xg)[i];
        int s4 = (i & 7) * 4;
        float4 wv = ((const float4*)n1w)[i];
        float4 bv = ((const float4*)n1b)[i];
        float h0 = (xv.x * (1.0f + sAtt[s4])     - mu) * rs * wv.x + bv.x;
        float h1 = (xv.y * (1.0f + sAtt[s4 + 1]) - mu) * rs * wv.y + bv.y;
        float h2 = (xv.z * (1.0f + sAtt[s4 + 2]) - mu) * rs * wv.z + bv.z;
        float h3 = (xv.w * (1.0f + sAtt[s4 + 3]) - mu) * rs * wv.w + bv.w;
        size_t base = (size_t)p * CHW + (size_t)i * 4;
        ((float4*)(hF + (size_t)p * CHW))[i] = make_float4(h0, h1, h2, h3);
        hB[base + 0] = (__bf16)h0; hB[base + 1] = (__bf16)h1;
        hB[base + 2] = (__bf16)h2; hB[base + 3] = (__bf16)h3;
    }
}

// ---------------------------------------------------------------- post-MLP: LN2 + pool + L2-normalize
__global__ void k_postmlp(const float* __restrict__ y2,
                          const float* __restrict__ n2w, const float* __restrict__ n2b,
                          float* __restrict__ pooledF, __bf16* __restrict__ pooledB,
                          float* __restrict__ feat) {
    int p = blockIdx.x, tid = threadIdx.x;
    const float* yp = y2 + (size_t)p * CHW;
    __shared__ float red[256], red2[256];

    float sum = 0.0f, sq = 0.0f;
    for (int i = tid; i < CHW / 4; i += 256) {
        float4 v = ((const float4*)yp)[i];
        sum += v.x + v.y + v.z + v.w;
        sq  += v.x * v.x + v.y * v.y + v.z * v.z + v.w * v.w;
    }
    red[tid] = sum; red2[tid] = sq;
    __syncthreads();
    for (int off = 128; off > 0; off >>= 1) {
        if (tid < off) { red[tid] += red[tid + off]; red2[tid] += red2[tid + off]; }
        __syncthreads();
    }
    float mu  = red[0] * (1.0f / CHW);
    float var = red2[0] * (1.0f / CHW) - mu * mu;
    float rs  = rsqrtf(var + 1e-5f);
    __syncthreads();

    float myp[8];
#pragma unroll
    for (int cc = 0; cc < 8; ++cc) {
        int c = tid + cc * 256;
        const float4* row = (const float4*)(yp + (size_t)c * S_DIM);
        const float4* wr  = (const float4*)(n2w + (size_t)c * S_DIM);
        const float4* br  = (const float4*)(n2b + (size_t)c * S_DIM);
        float a = 0.0f;
#pragma unroll
        for (int t = 0; t < 8; ++t) {
            float4 v = row[t], w = wr[t], bb = br[t];
            a += (v.x - mu) * rs * w.x + bb.x;
            a += (v.y - mu) * rs * w.y + bb.y;
            a += (v.z - mu) * rs * w.z + bb.z;
            a += (v.w - mu) * rs * w.w + bb.w;
        }
        a *= (1.0f / S_DIM);
        myp[cc] = a;
        pooledF[(size_t)p * C_DIM + c] = a;
        pooledB[(size_t)p * C_DIM + c] = (__bf16)a;
    }

    float s2 = 0.0f;
#pragma unroll
    for (int cc = 0; cc < 8; ++cc) s2 += myp[cc] * myp[cc];
    red[tid] = s2;
    __syncthreads();
    for (int off = 128; off > 0; off >>= 1) {
        if (tid < off) red[tid] += red[tid + off];
        __syncthreads();
    }
    float inv = 1.0f / fmaxf(sqrtf(red[0]), 1e-12f);
#pragma unroll
    for (int cc = 0; cc < 8; ++cc)
        feat[(size_t)p * C_DIM + tid + cc * 256] = myp[cc] * inv;
}

// ---------------------------------------------------------------- fc: logits = pooled @ fc_w^T + b  (WMMA)
__global__ void k_fc(const __bf16* __restrict__ pooledB, const __bf16* __restrict__ fcT,
                     const float* __restrict__ fcb, float* __restrict__ outFc) {
    int wv   = (blockIdx.x * blockDim.x + threadIdx.x) >> 5;  // 0..767
    int lane = threadIdx.x & 31;
    int nlo = lane & 15, hf = lane >> 4;
    int mt = wv / 48, nt = wv % 48;
    int m0 = mt * 16, n0 = nt * 16;

    v8f acc = {};
    for (int k0 = 0; k0 < C_DIM; k0 += 32) {
        V16 a, b;
        const __bf16* ar = pooledB + (size_t)(m0 + nlo) * C_DIM + k0 + 8 * hf;
        a.q[0] = *(const uint4*)ar;
        a.q[1] = *(const uint4*)(ar + 16);
        const unsigned short* br = (const unsigned short*)fcT;
#pragma unroll
        for (int s = 0; s < 16; ++s) {
            int kk = (s < 8) ? (8 * hf + s) : (16 + 8 * hf + (s - 8));
            b.h[s] = br[(size_t)(k0 + kk) * NCLS_PAD + n0 + nlo];
        }
        acc = wmma_bf16(a, b, acc);
    }

    int j = n0 + nlo;
    if (j < NCLS) {
        float bv = fcb[j];
#pragma unroll
        for (int r = 0; r < 8; ++r) {
            int pi = m0 + r + 8 * hf;
            outFc[(size_t)pi * NCLS + j] = acc[r] + bv;
        }
    }
}

// ---------------------------------------------------------------- launcher
extern "C" void kernel_launch(void* const* d_in, const int* in_sizes, int n_in,
                              void* d_out, int out_size, void* d_ws, size_t ws_size,
                              hipStream_t stream) {
    const float* x       = (const float*)d_in[0];
    const float* theta_w = (const float*)d_in[1];
    const float* theta_b = (const float*)d_in[2];
    const float* phi_w   = (const float*)d_in[3];
    const float* phi_b   = (const float*)d_in[4];
    const float* conv1_w = (const float*)d_in[5];
    const float* conv2_w = (const float*)d_in[6];
    const float* norm1_w = (const float*)d_in[7];
    const float* norm1_b = (const float*)d_in[8];
    const float* norm2_w = (const float*)d_in[9];
    const float* norm2_b = (const float*)d_in[10];
    const float* fc_w    = (const float*)d_in[11];
    const float* fc_b    = (const float*)d_in[12];
    (void)in_sizes; (void)n_in; (void)out_size; (void)ws_size;

    float* out     = (float*)d_out;
    float* outFeat = out;                                   // [256,2048]
    float* outFc   = out + (size_t)P_PAIR * C_DIM;          // [256,751]
    float* outAtt  = outFc + (size_t)P_PAIR * NCLS;         // [256,32]

    char* w = (char*)d_ws;
    auto carve = [&](size_t bytes) -> void* {
        void* p = (void*)w;
        w += (bytes + 255) & ~(size_t)255;
        return p;
    };
    __bf16* Wt      = (__bf16*)carve((size_t)C_DIM * C_DIM * 2);
    __bf16* Wp      = (__bf16*)carve((size_t)C_DIM * C_DIM * 2);
    __bf16* W1      = (__bf16*)carve((size_t)C_DIM * C_DIM * 2);
    __bf16* W2      = (__bf16*)carve((size_t)C_DIM * C_DIM * 2);
    __bf16* Xb      = (__bf16*)carve((size_t)N_IMG * CHW * 2);
    __bf16* TP      = (__bf16*)carve((size_t)N_IMG * CHW * 2);
    __bf16* TGt     = (__bf16*)carve((size_t)32 * CHW * 2);     // theta^T, galleries
    __bf16* fcT     = (__bf16*)carve((size_t)C_DIM * NCLS_PAD * 2);
    float*  hF      = (float*)carve((size_t)P_PAIR * CHW * 4);
    __bf16* hB      = (__bf16*)carve((size_t)P_PAIR * CHW * 2);
    __bf16* h1B     = (__bf16*)carve((size_t)P_PAIR * CHW * 2);
    float*  pooledF = (float*)carve((size_t)P_PAIR * C_DIM * 4);
    __bf16* pooledB = (__bf16*)carve((size_t)P_PAIR * C_DIM * 2);

    // 1. precision conversion
    k_f32_to_bf16<<<1024, 256, 0, stream>>>(theta_w, Wt, C_DIM * C_DIM);
    k_f32_to_bf16<<<1024, 256, 0, stream>>>(phi_w,   Wp, C_DIM * C_DIM);
    k_f32_to_bf16<<<1024, 256, 0, stream>>>(conv1_w, W1, C_DIM * C_DIM);
    k_f32_to_bf16<<<1024, 256, 0, stream>>>(conv2_w, W2, C_DIM * C_DIM);
    k_f32_to_bf16<<<1024, 256, 0, stream>>>(x,       Xb, N_IMG * CHW);
    k_fc_transpose<<<1024, 256, 0, stream>>>(fc_w, fcT);

    // 2. theta (galleries) / phi (queries): 40 images x 4 blocks
    k_gemm<0><<<160, 256, 0, stream>>>(Wt, Wp, theta_b, phi_b, Xb, TP, nullptr, TGt);

    // 3. attention map: one wave per pair
    k_attention<<<32, 256, 0, stream>>>(TP, TGt, outAtt);

    // 4. residual + LN1 -> h
    k_premlp<<<P_PAIR, 256, 0, stream>>>(x, outAtt, norm1_w, norm1_b, hF, hB);

    // 5. MLP: conv1(relu) then conv2 (+residual into hF): 256 pairs x 4 blocks
    k_gemm<1><<<1024, 256, 0, stream>>>(W1, nullptr, nullptr, nullptr, hB, h1B, nullptr, nullptr);
    k_gemm<2><<<1024, 256, 0, stream>>>(W2, nullptr, nullptr, nullptr, h1B, nullptr, hF, nullptr);

    // 6. LN2 + pool + normalize -> feat, pooled
    k_postmlp<<<P_PAIR, 256, 0, stream>>>(hF, norm2_w, norm2_b, pooledF, pooledB, outFeat);

    // 7. classifier: 16 x 48 tiles = 768 waves
    k_fc<<<96, 256, 0, stream>>>(pooledB, fcT, fc_b, outFc);
}
